// TitanBlock_45947560132655
// MI455X (gfx1250) — matrix operand order repaired and checked by reading, not compiled, not run on hardware
//
#include <hip/hip_runtime.h>

#define DEVF __device__ __forceinline__

typedef __attribute__((ext_vector_type(16))) __bf16 v16bf;
typedef __attribute__((ext_vector_type(8)))  __bf16 v8bf;
typedef __attribute__((ext_vector_type(8)))  float  v8f;

static constexpr int Bc = 2, Tc = 2048, Cc = 1024, Hc = 16, Dc = 64;
static constexpr int Mc = Bc * Tc; // 4096 tokens

DEVF v8f vzero8() { v8f z = {0.f,0.f,0.f,0.f,0.f,0.f,0.f,0.f}; return z; }

// ---- CDNA5 WMMA fragment packing (per CDNA5 ISA §7.12.2) ----
// A 16x32 bf16: lane m (0-15) holds M=m, K = {0..7, 16..23}; lane m+16 holds K = {8..15, 24..31}
DEVF v16bf load_a_frag(const __bf16* base, int ld, int lane) {
  const int m  = lane & 15;
  const int kb = (lane >> 4) << 3;            // 0 or 8
  const __bf16* p = base + (size_t)m * ld + kb;
  v8bf lo = *reinterpret_cast<const v8bf*>(p);        // K = kb .. kb+7
  v8bf hi = *reinterpret_cast<const v8bf*>(p + 16);   // K = kb+16 .. kb+23
  v16bf r;
#pragma unroll
  for (int e = 0; e < 8; ++e) { r[e] = lo[e]; r[e + 8] = hi[e]; }
  return r;
}

// B 32x16 bf16: lane n (0-15) holds column n, K = {0..7, 16..23}; lane n+16 holds column n, K = {8..15, 24..31}
// base points at transposed weight [N][K] (column n contiguous in K)
DEVF v16bf load_b_frag(const __bf16* base, int ld, int lane) {
  const int n  = lane & 15;
  const int kb = (lane >> 4) << 3;            // 0 or 8
  const __bf16* p = base + (size_t)n * ld + kb;
  v8bf lo = *reinterpret_cast<const v8bf*>(p);
  v8bf hi = *reinterpret_cast<const v8bf*>(p + 16);
  v16bf r;
#pragma unroll
  for (int e = 0; e < 8; ++e) { r[e] = lo[e]; r[e + 8] = hi[e]; }
  return r;
}

DEVF v8f wmma_bf16(v16bf a, v16bf b, v8f c) {
  // (neg_a, A, neg_b, B, c_mod, C, reuse_a, reuse_b)
  return __builtin_amdgcn_wmma_f32_16x16x32_bf16(false, a, false, b, (short)0, c, false, false);
}

// ---- weight fp32 [K][N] -> bf16 [N][K] (LDS-tiled transpose) ----
__global__ __launch_bounds__(256) void transpose_f32_to_bf16(
    const float* __restrict__ in, __bf16* __restrict__ out, int K, int N) {
  __shared__ float tile[32][33];
  const int kb = blockIdx.x * 32, nb = blockIdx.y * 32;
  const int tx = threadIdx.x & 31, ty = threadIdx.x >> 5;
#pragma unroll
  for (int r = ty; r < 32; r += 8)
    tile[r][tx] = in[(size_t)(kb + r) * N + (nb + tx)];
  __syncthreads();
#pragma unroll
  for (int r = ty; r < 32; r += 8)
    out[(size_t)(nb + r) * K + (kb + tx)] = (__bf16)tile[tx][r];
}

// ---- fused LayerNorm (fp32 in) -> bf16 out ----
__global__ __launch_bounds__(256) void layernorm_to_bf16(
    const float* __restrict__ x, const float* __restrict__ w,
    const float* __restrict__ bb, __bf16* __restrict__ out) {
  const int row = blockIdx.x;
  const float* xr = x + (size_t)row * Cc;
  float s = 0.f, s2 = 0.f;
  for (int i = threadIdx.x; i < Cc; i += 256) { float v = xr[i]; s += v; s2 += v * v; }
#pragma unroll
  for (int off = 16; off > 0; off >>= 1) { s += __shfl_xor(s, off); s2 += __shfl_xor(s2, off); }
  __shared__ float rs[8], rs2[8];
  const int wave = threadIdx.x >> 5, lane = threadIdx.x & 31;
  if (lane == 0) { rs[wave] = s; rs2[wave] = s2; }
  __syncthreads();
  s = 0.f; s2 = 0.f;
#pragma unroll
  for (int i = 0; i < 8; ++i) { s += rs[i]; s2 += rs2[i]; }
  const float mean = s / Cc;
  const float var  = s2 / Cc - mean * mean;
  const float inv  = rsqrtf(var + 1e-5f);
  for (int i = threadIdx.x; i < Cc; i += 256)
    out[(size_t)row * Cc + i] = (__bf16)(((xr[i] - mean) * inv) * w[i] + bb[i]);
}

// ---- generic bf16 WMMA GEMM: out[M,N] = A[M,K] @ Wt[N,K]^T + bias (+GELU) (+res) ----
// Block = 256 threads = 8 waves (2 along M x 4 along N); block tile 128x256.
// Wave tile 64x64: 4x4 accumulators -> 16 WMMAs per K-step against 8 fragment loads.
template<bool GELU_ACT, bool ADD_RES, bool OUT_F32>
__global__ __launch_bounds__(256) void gemm_bf16_wmma(
    const __bf16* __restrict__ A, const __bf16* __restrict__ Wt,
    const float* __restrict__ bias, const float* __restrict__ res,
    void* __restrict__ outv, int M, int N, int K) {
  const int lane = threadIdx.x & 31;
  const int wave = threadIdx.x >> 5;
  const int row0 = blockIdx.x * 128 + (wave & 1) * 64;   // 2 waves along M
  const int col0 = blockIdx.y * 256 + (wave >> 1) * 64;  // 4 waves along N
  v8f acc[4][4];
#pragma unroll
  for (int i = 0; i < 4; ++i)
#pragma unroll
    for (int j = 0; j < 4; ++j) acc[i][j] = vzero8();

  for (int k0 = 0; k0 < K; k0 += 32) {
    v16bf a[4], b[4];
#pragma unroll
    for (int i = 0; i < 4; ++i)
      a[i] = load_a_frag(A + (size_t)(row0 + 16 * i) * K + k0, K, lane);
#pragma unroll
    for (int j = 0; j < 4; ++j)
      b[j] = load_b_frag(Wt + (size_t)(col0 + 16 * j) * K + k0, K, lane);
#pragma unroll
    for (int i = 0; i < 4; ++i)
#pragma unroll
      for (int j = 0; j < 4; ++j)
        acc[i][j] = wmma_bf16(a[i], b[j], acc[i][j]);
  }
  // C/D layout: element (m,n) -> lane = n + 16*(m>=8), vgpr r = m&7
  const int nn = lane & 15;
  const int mb = (lane >> 4) << 3;
#pragma unroll
  for (int j = 0; j < 4; ++j) {
    const int col = col0 + 16 * j + nn;
    const float bv = bias[col];
#pragma unroll
    for (int i = 0; i < 4; ++i) {
#pragma unroll
      for (int r = 0; r < 8; ++r) {
        const int row = row0 + 16 * i + mb + r;
        float v = acc[i][j][r] + bv;
        if (GELU_ACT) v = 0.5f * v * (1.f + erff(v * 0.70710678118654752f));
        if (ADD_RES) v += res[(size_t)row * N + col];
        if (OUT_F32) ((float*)outv)[(size_t)row * N + col] = v;
        else ((__bf16*)outv)[(size_t)row * N + col] = (__bf16)v;
      }
    }
  }
}

// ---- repack qkv [M][3C] bf16 -> Q[bh][T][D], K[bh][T][D], Vt[bh][D][T] ----
__global__ __launch_bounds__(256) void repack_qkv(
    const __bf16* __restrict__ qkv, __bf16* __restrict__ Q,
    __bf16* __restrict__ Kd, __bf16* __restrict__ Vt) {
  const size_t idx = (size_t)blockIdx.x * 256 + threadIdx.x; // ((b*T+t)*H+h)*D+d
  if (idx >= (size_t)Mc * Cc) return;
  const int d = (int)(idx & (Dc - 1));
  const int h = (int)((idx >> 6) & (Hc - 1));
  const int t = (int)((idx >> 10) & (Tc - 1));
  const int b = (int)(idx >> 21);
  const size_t row = (size_t)b * Tc + t;
  const int col = h * Dc + d;
  const int bh = b * Hc + h;
  Q [((size_t)bh * Tc + t) * Dc + d] = qkv[row * (3 * Cc) + col];
  Kd[((size_t)bh * Tc + t) * Dc + d] = qkv[row * (3 * Cc) + Cc + col];
  Vt[((size_t)bh * Dc + d) * Tc + t] = qkv[row * (3 * Cc) + 2 * Cc + col];
}

// ---- flash-style strided-causal attention, one wave = 16 query rows ----
__global__ __launch_bounds__(256) void attn_strided(
    const __bf16* __restrict__ Q, const __bf16* __restrict__ Kd,
    const __bf16* __restrict__ Vt, __bf16* __restrict__ Y) {
  const int lane = threadIdx.x & 31;
  const int wave = threadIdx.x >> 5;
  const int bh = blockIdx.y;
  const int b = bh >> 4, h = bh & 15;
  const int q0 = blockIdx.x * 128 + wave * 16;
  const __bf16* Qb = Q  + (size_t)bh * Tc * Dc;
  const __bf16* Kb = Kd + (size_t)bh * Tc * Dc;
  const __bf16* Vb = Vt + (size_t)bh * Dc * Tc;
  __shared__ __align__(16) __bf16 plds[8][16 * 32];   // per-wave P bounce buffer
  __bf16* pl = plds[wave];

  const v16bf qa0 = load_a_frag(Qb + (size_t)q0 * Dc,      Dc, lane);
  const v16bf qa1 = load_a_frag(Qb + (size_t)q0 * Dc + 32, Dc, lane);
  v8f o0 = vzero8(), o1 = vzero8(), o2 = vzero8(), o3 = vzero8();
  float rmax[8], rsum[8];
#pragma unroll
  for (int r = 0; r < 8; ++r) { rmax[r] = -1e30f; rsum[r] = 0.f; }
  const int nn = lane & 15;
  const int mb = (lane >> 4) << 3;
  const float scale = 0.125f; // 1/sqrt(64)

  for (int k0 = 0; k0 <= q0 + 15; k0 += 32) {
    // S = Q(16x64) . K^T(64x32), split into two 16x16 tiles
    v8f s0 = vzero8(), s1 = vzero8();
    s0 = wmma_bf16(qa0, load_b_frag(Kb + (size_t)k0 * Dc,             Dc, lane), s0);
    s0 = wmma_bf16(qa1, load_b_frag(Kb + (size_t)k0 * Dc + 32,        Dc, lane), s0);
    s1 = wmma_bf16(qa0, load_b_frag(Kb + (size_t)(k0 + 16) * Dc,      Dc, lane), s1);
    s1 = wmma_bf16(qa1, load_b_frag(Kb + (size_t)(k0 + 16) * Dc + 32, Dc, lane), s1);

    const int j0 = k0 + nn, j1 = j0 + 16;
    float p0[8], p1[8];
#pragma unroll
    for (int r = 0; r < 8; ++r) {
      const int i = q0 + mb + r;
      // allowed: j<=i && (j >= i-4 || j % 2 == 0)
      float v0 = (j0 <= i && (j0 >= i - 4 || (j0 & 1) == 0)) ? s0[r] * scale : -1e30f;
      float v1 = (j1 <= i && (j1 >= i - 4 || (j1 & 1) == 0)) ? s1[r] * scale : -1e30f;
      float mx = fmaxf(v0, v1);
#pragma unroll
      for (int off = 1; off < 16; off <<= 1) mx = fmaxf(mx, __shfl_xor(mx, off));
      const float nm  = fmaxf(rmax[r], mx);
      const float fac = __expf(rmax[r] - nm);
      const float e0 = __expf(v0 - nm), e1 = __expf(v1 - nm);
      float ps = e0 + e1;
#pragma unroll
      for (int off = 1; off < 16; off <<= 1) ps += __shfl_xor(ps, off);
      rsum[r] = rsum[r] * fac + ps;
      rmax[r] = nm;
      p0[r] = e0; p1[r] = e1;
      o0[r] *= fac; o1[r] *= fac; o2[r] *= fac; o3[r] *= fac;
    }
    // C-layout P -> LDS -> A-layout P (same-wave transpose; DS ops are in-order per wave)
    __builtin_amdgcn_wave_barrier();
#pragma unroll
    for (int r = 0; r < 8; ++r) {
      pl[(mb + r) * 32 + nn]      = (__bf16)p0[r];
      pl[(mb + r) * 32 + nn + 16] = (__bf16)p1[r];
    }
    __builtin_amdgcn_wave_barrier();
    asm volatile("s_wait_dscnt 0" ::: "memory");
    const v16bf pa = load_a_frag(pl, 32, lane);
    __builtin_amdgcn_wave_barrier();
    // O(16x64) += P(16x32) . V(32x64); Vt is [D][T] so B-lanes read contiguous K
    o0 = wmma_bf16(pa, load_b_frag(Vb + (size_t) 0 * Tc + k0, Tc, lane), o0);
    o1 = wmma_bf16(pa, load_b_frag(Vb + (size_t)16 * Tc + k0, Tc, lane), o1);
    o2 = wmma_bf16(pa, load_b_frag(Vb + (size_t)32 * Tc + k0, Tc, lane), o2);
    o3 = wmma_bf16(pa, load_b_frag(Vb + (size_t)48 * Tc + k0, Tc, lane), o3);
  }
#pragma unroll
  for (int r = 0; r < 8; ++r) {
    const float inv = 1.f / rsum[r];
    __bf16* yr = Y + ((size_t)(b * Tc + q0 + mb + r)) * Cc + h * Dc;
    yr[nn]      = (__bf16)(o0[r] * inv);
    yr[16 + nn] = (__bf16)(o1[r] * inv);
    yr[32 + nn] = (__bf16)(o2[r] * inv);
    yr[48 + nn] = (__bf16)(o3[r] * inv);
  }
}

extern "C" void kernel_launch(void* const* d_in, const int* in_sizes, int n_in,
                              void* d_out, int out_size, void* d_ws, size_t ws_size,
                              hipStream_t stream) {
  (void)in_sizes; (void)n_in; (void)out_size; (void)ws_size;
  const float* x           = (const float*)d_in[0];
  const float* ln1_w       = (const float*)d_in[1];
  const float* ln1_b       = (const float*)d_in[2];
  const float* c_attn_w    = (const float*)d_in[3];
  const float* c_attn_b    = (const float*)d_in[4];
  const float* attn_proj_w = (const float*)d_in[5];
  const float* attn_proj_b = (const float*)d_in[6];
  const float* ln2_w       = (const float*)d_in[7];
  const float* ln2_b       = (const float*)d_in[8];
  const float* fc_w        = (const float*)d_in[9];
  const float* fc_b        = (const float*)d_in[10];
  const float* mlp_proj_w  = (const float*)d_in[11];
  const float* mlp_proj_b  = (const float*)d_in[12];

  char* ws = (char*)d_ws;
  size_t off = 0;
  auto alloc = [&](size_t bytes) -> void* {
    void* p = ws + off; off += (bytes + 255) & ~(size_t)255; return p;
  };
  const size_t M = Mc, C = Cc;
  __bf16* h1    = (__bf16*)alloc(M * C * 2);              // ln1(x) bf16
  __bf16* Wqkv  = (__bf16*)alloc((size_t)3 * C * C * 2);  // [3C][C]
  __bf16* Wproj = (__bf16*)alloc((size_t)C * C * 2);      // [C][C]
  __bf16* Wfc   = (__bf16*)alloc((size_t)4 * C * C * 2);  // [4C][C]
  __bf16* Wmlp  = (__bf16*)alloc((size_t)4 * C * C * 2);  // [C][4C]
  __bf16* qkv   = (__bf16*)alloc(M * 3 * C * 2);
  __bf16* Qb    = (__bf16*)alloc(M * C * 2);
  __bf16* Kb    = (__bf16*)alloc(M * C * 2);
  __bf16* Vtb   = (__bf16*)alloc(M * C * 2);
  __bf16* Yb    = (__bf16*)alloc(M * C * 2);
  float*  x1    = (float*)alloc(M * C * 4);
  __bf16* h2    = (__bf16*)alloc(M * C * 2);
  __bf16* hfc   = (__bf16*)alloc(M * 4 * C * 2);

  dim3 blk(256);
  // 1) weights -> bf16 transposed [N][K]
  transpose_f32_to_bf16<<<dim3(C / 32, 3 * C / 32), blk, 0, stream>>>(c_attn_w, Wqkv, C, 3 * C);
  transpose_f32_to_bf16<<<dim3(C / 32, C / 32),     blk, 0, stream>>>(attn_proj_w, Wproj, C, C);
  transpose_f32_to_bf16<<<dim3(C / 32, 4 * C / 32), blk, 0, stream>>>(fc_w, Wfc, C, 4 * C);
  transpose_f32_to_bf16<<<dim3(4 * C / 32, C / 32), blk, 0, stream>>>(mlp_proj_w, Wmlp, 4 * C, C);
  // 2) LN1
  layernorm_to_bf16<<<dim3((unsigned)M), blk, 0, stream>>>(x, ln1_w, ln1_b, h1);
  // 3) qkv = h1 @ Wqkv + b  (bf16 out)
  gemm_bf16_wmma<false, false, false><<<dim3(M / 128, 3 * C / 256), blk, 0, stream>>>(
      h1, Wqkv, c_attn_b, nullptr, qkv, (int)M, 3 * Cc, Cc);
  // 4) repack for attention
  repack_qkv<<<dim3((unsigned)((M * C) / 256)), blk, 0, stream>>>(qkv, Qb, Kb, Vtb);
  // 5) strided-causal flash attention
  attn_strided<<<dim3(Tc / 128, Bc * Hc), blk, 0, stream>>>(Qb, Kb, Vtb, Yb);
  // 6) x1 = x + Y @ Wproj + b  (f32 out)
  gemm_bf16_wmma<false, true, true><<<dim3(M / 128, C / 256), blk, 0, stream>>>(
      Yb, Wproj, attn_proj_b, x, x1, (int)M, Cc, Cc);
  // 7) LN2
  layernorm_to_bf16<<<dim3((unsigned)M), blk, 0, stream>>>(x1, ln2_w, ln2_b, h2);
  // 8) hfc = gelu(h2 @ Wfc + b) (bf16 out)
  gemm_bf16_wmma<true, false, false><<<dim3(M / 128, 4 * C / 256), blk, 0, stream>>>(
      h2, Wfc, fc_b, nullptr, hfc, (int)M, 4 * Cc, Cc);
  // 9) out = x1 + hfc @ Wmlp + b (f32 out)
  gemm_bf16_wmma<false, true, true><<<dim3(M / 128, C / 256), blk, 0, stream>>>(
      hfc, Wmlp, mlp_proj_b, x1, (float*)d_out, (int)M, Cc, 4 * Cc);
}
